// bg_G_78812649881892
// MI455X (gfx1250) — matrix operand order repaired and checked by reading, not compiled, not run on hardware
//
#include <hip/hip_runtime.h>

typedef __attribute__((ext_vector_type(16))) _Float16 v16h;
typedef __attribute__((ext_vector_type(8)))  _Float16 v8h;
typedef __attribute__((ext_vector_type(8)))  float    v8f;

#define B_   8
#define H_   64
#define W_   128
#define S_   35
#define SH_  256
#define SW_  512
#define SD_  64
#define NH_  128
#define CIN_ 256

// ---------------------------------------------------------------------------
// Weight swizzle: f32 [Cout][CinSrc][3][3] -> f16 fragments laid out so that
// each lane of a wave reads its 16-half B fragment contiguously.
// dst[((nt*KC + kc)*32 + lane)*16 + i] = w[cout=nt*16+(lane&15)][tap][cin]
// with kc -> (tap, c0), cin = c0 + (i>>3)*16 + (lane>>4)*8 + (i&7)  (zero-padded past CinSrc)
// ---------------------------------------------------------------------------
__global__ void __launch_bounds__(256)
swizzle_w_kernel(const float* __restrict__ w, _Float16* __restrict__ dst,
                 int Cout, int CinSrc, int Cinp) {
    int idx = blockIdx.x * 256 + threadIdx.x;
    int total = Cout * 9 * Cinp;
    if (idx >= total) return;
    int KC   = 9 * Cinp / 32;
    int i    = idx & 15;
    int lane = (idx >> 4) & 31;
    int t    = idx >> 9;
    int kc   = t % KC;
    int nt   = t / KC;
    int cout = nt * 16 + (lane & 15);
    int CC   = Cinp / 32;
    int tap  = kc / CC;
    int c0   = (kc % CC) * 32;
    int cin  = c0 + (i & 8) * 2 + (lane >> 4) * 8 + (i & 7);
    float v = 0.f;
    if (cin < CinSrc)
        v = w[(((size_t)cout * CinSrc + cin) * 3 + tap / 3) * 3 + (tap % 3)];
    dst[idx] = (_Float16)v;
}

// ---------------------------------------------------------------------------
// Generic 3x3 SAME conv as WMMA GEMM.
//   M = 16 pixels along W (one row segment), N = 16 Cout per tile,
//   K = 9*CINP reduced in chunks of 32 (f16).
// AIN16:  input is f16 NHWC -> stage patch with global_load_async_to_lds_b128
//         (per-lane 16B gather, no VGPR roundtrip, no cvt); OOB halo cells are
//         zero-filled with ds stores; fence = s_wait_asynccnt 0 + barrier.
// !AIN16: input is f32 NCHW -> VGPR load + cvt to f16 in LDS.
// AOUT16: write f16 NHWC (coalesced across lanes); else f32 NCHW.
// DUAL:   two weight sets (gamma/beta) sharing the A operand; epilogue fuses
//         lrelu(nx*(1+gamma)+beta) and writes f32 NCHW.
// ACT: 0=none, 1=lrelu(0.2), 2=relu
// ---------------------------------------------------------------------------
template <int CINP, int COUT, int ACT, bool DUAL, bool AIN16, bool AOUT16>
__global__ void __launch_bounds__(256)
conv3x3_wmma(const void* __restrict__ xin,
             const _Float16* __restrict__ wa,
             const _Float16* __restrict__ wb,
             const float* __restrict__ bias_a,
             const float* __restrict__ bias_b,
             const float* __restrict__ nx,
             void* __restrict__ outv) {
    extern __shared__ _Float16 lds[];          // 3*18*(CINP+8) halfs
    constexpr int SP = CINP + 8;               // padded row stride (halfs)
    constexpr int KC = 9 * CINP / 32;          // K chunks
    constexpr int NT = COUT / 16;              // N tiles
    constexpr int NTPW = (NT + 7) / 8;         // N tiles per wave

    const int Wt = W_ >> 4;
    int blk = blockIdx.x;
    int x0  = (blk % Wt) << 4;
    int y   = (blk / Wt) % H_;
    int b   = blk / (Wt * H_);

    int tid = threadIdx.x;

    // ---- Phase 1: stage 3x18xCINP input patch into LDS as f16 -------------
    if constexpr (AIN16) {
        const _Float16* x16 = (const _Float16*)xin;   // NHWC [B,H,W,CINP]
        const int blocks = 3 * 18 * (CINP / 8);       // 16B blocks
        unsigned ldsbase = (unsigned)__builtin_amdgcn_groupstaticsize();
        v8h zz = {};
        for (int e = tid; e < blocks; e += 256) {
            int c8  = e % (CINP / 8);
            int rc  = e / (CINP / 8);
            int col = rc % 18;
            int r   = rc / 18;
            int c   = c8 * 8;
            int yy = y + r - 1;
            int xx = x0 + col - 1;
            unsigned loff = ldsbase + (unsigned)(((r * 18 + col) * SP + c) * 2);
            if ((unsigned)yy < (unsigned)H_ && (unsigned)xx < (unsigned)W_) {
                unsigned long long ga = (unsigned long long)(const void*)
                    (x16 + ((size_t)(b * H_ + yy) * W_ + xx) * CINP + c);
                asm volatile("global_load_async_to_lds_b128 %0, %1, off"
                             :: "v"(loff), "v"(ga) : "memory");
            } else {
                *(v8h*)(lds + ((r * 18 + col) * SP + c)) = zz;
            }
        }
        asm volatile("s_wait_asynccnt 0" ::: "memory");
        __syncthreads();
    } else {
        const float* x = (const float*)xin;           // NCHW [B,CINP,H,W]
        const int total = 3 * 18 * CINP;
        for (int e = tid; e < total; e += 256) {
            int c   = e / 54;
            int rc  = e % 54;
            int r   = rc / 18;
            int col = rc % 18;
            int yy = y + r - 1;
            int xx = x0 + col - 1;
            float v = 0.f;
            if ((unsigned)yy < (unsigned)H_ && (unsigned)xx < (unsigned)W_)
                v = x[(((size_t)b * CINP + c) * H_ + yy) * W_ + xx];
            lds[(r * 18 + col) * SP + c] = (_Float16)v;
        }
        __syncthreads();
    }

    int lane = tid & 31;
    int wave = tid >> 5;
    int mrow = lane & 15;   // A: M row ; B/D: N column
    int hig  = lane >> 4;   // lane-group select

    if (NT >= 8 || wave < NT) {
        v8f zero = {};
        v8f acc[NTPW];
        v8f accB[DUAL ? NTPW : 1];
        #pragma unroll
        for (int j = 0; j < NTPW; ++j) { acc[j] = zero; if (DUAL) accB[j] = zero; }

        for (int kc = 0; kc < KC; ++kc) {
            constexpr int CC = CINP / 32;
            int tap = kc / CC;
            int c0  = (kc % CC) * 32;
            int r   = tap / 3;
            int tx  = tap % 3;
            // A fragment: lane (hig,mrow): elems 0..7 = K hig*8+0..7,
            //             elems 8..15 = K 16+hig*8+0..7 (ISA 16-bit 16x32 layout)
            int base = (r * 18 + (mrow + tx)) * SP + c0 + hig * 8;
            v8h lo = *(const v8h*)(lds + base);
            v8h hi = *(const v8h*)(lds + base + 16);
            v16h a;
            #pragma unroll
            for (int i = 0; i < 8; ++i) { a[i] = lo[i]; a[8 + i] = hi[i]; }

            #pragma unroll
            for (int j = 0; j < NTPW; ++j) {
                int nt = wave + j * 8;
                if (NT >= 8 || nt < NT) {
                    const _Float16* pb = wa + (((size_t)nt * KC + kc) * 32 + lane) * 16;
                    v16h bf = *(const v16h*)pb;
                    acc[j] = __builtin_amdgcn_wmma_f32_16x16x32_f16(
                        false, a, false, bf, (short)0, acc[j], false, false);
                    if (DUAL) {
                        const _Float16* pb2 = wb + (((size_t)nt * KC + kc) * 32 + lane) * 16;
                        v16h bf2 = *(const v16h*)pb2;
                        accB[j] = __builtin_amdgcn_wmma_f32_16x16x32_f16(
                            false, a, false, bf2, (short)0, accB[j], false, false);
                    }
                }
            }
        }

        // ---- Epilogue (C/D layout: N = lane&15, M = v + 8*(lane>>4)) ------
        #pragma unroll
        for (int j = 0; j < NTPW; ++j) {
            int nt = wave + j * 8;
            if (NT >= 8 || nt < NT) {
                int cout = nt * 16 + mrow;
                float bv  = bias_a[cout];
                float bv2 = DUAL ? bias_b[cout] : 0.f;
                size_t obase_nchw = (((size_t)b * COUT + cout) * H_ + y) * W_ + x0;
                size_t obase_nhwc = ((size_t)(b * H_ + y) * W_ + x0) * COUT + cout;
                #pragma unroll
                for (int v = 0; v < 8; ++v) {
                    int m = v + 8 * hig;
                    float val = acc[j][v] + bv;
                    if (DUAL) {
                        float g  = val;
                        float bt = accB[j][v] + bv2;
                        float nv = nx[obase_nchw + m];
                        float o  = nv * (1.f + g) + bt;
                        val = (o > 0.f) ? o : 0.2f * o;
                    } else if (ACT == 1) {
                        val = (val > 0.f) ? val : 0.2f * val;
                    } else if (ACT == 2) {
                        val = (val > 0.f) ? val : 0.f;
                    }
                    if constexpr (AOUT16) {
                        ((_Float16*)outv)[obase_nhwc + (size_t)m * COUT] = (_Float16)val;
                    } else {
                        ((float*)outv)[obase_nchw + m] = val;
                    }
                }
            }
        }
    }
}

// ---------------------------------------------------------------------------
// Per-(batch,class) masked mean of featmap over spatial -> codes [B,S,64].
// mb = (seg*bg*(1-mask) != 0) at 4x-decimated segmap resolution.
// ---------------------------------------------------------------------------
__global__ void __launch_bounds__(256)
stats_kernel(const float* __restrict__ feat, const float* __restrict__ seg,
             const float* __restrict__ bg, const float* __restrict__ mask,
             float* __restrict__ codes) {
    __shared__ float ssum[SD_];
    __shared__ float scnt;
    int b = blockIdx.x / S_;
    int s = blockIdx.x % S_;
    int tid = threadIdx.x;
    if (tid < SD_) ssum[tid] = 0.f;
    if (tid == 0) scnt = 0.f;
    __syncthreads();
    for (int p = tid; p < H_ * W_; p += 256) {
        int h = p / W_, w = p % W_;
        int hh = h * 4, ww = w * 4;
        float sv = seg[(((size_t)b * S_ + s) * SH_ + hh) * SW_ + ww];
        float bv = bg[((size_t)b * SH_ + hh) * SW_ + ww];
        float mv = mask[((size_t)b * SH_ + hh) * SW_ + ww];
        float ctx = sv * bv * (1.f - mv);
        if (ctx != 0.f) {
            atomicAdd(&scnt, 1.f);
            for (int f = 0; f < SD_; ++f)
                atomicAdd(&ssum[f], feat[(((size_t)b * SD_ + f) * H_ + h) * W_ + w]);
        }
    }
    __syncthreads();
    if (tid < SD_) {
        float cnt = scnt > 1.f ? scnt : 1.f;
        float keep = (s >= 24 && s <= 33) ? 0.f : 1.f;  // CLASS_SELECTED
        codes[((size_t)b * S_ + s) * SD_ + tid] = keep * ssum[tid] / cnt;
    }
}

// ---------------------------------------------------------------------------
// Build out_map as f16 NHWC [B,H,W,128]: ch 0..63 = codes[label]*bg_segmap,
// ch 64..98 = bg_segmap (one-hot * bg), ch 99..127 = 0 (pad for mlp conv).
// ---------------------------------------------------------------------------
__global__ void __launch_bounds__(256)
stylemap_kernel(const float* __restrict__ seg, const float* __restrict__ bg,
                const float* __restrict__ codes, _Float16* __restrict__ omap) {
    int p = blockIdx.x * 256 + threadIdx.x;
    if (p >= B_ * H_ * W_) return;
    int w = p % W_;
    int h = (p / W_) % H_;
    int b = p / (W_ * H_);
    int hh = h * 4, ww = w * 4;
    float bv = bg[((size_t)b * SH_ + hh) * SW_ + ww];
    int lab = 0; float sv = 0.f;
    for (int s = 0; s < S_; ++s) {
        float v = seg[(((size_t)b * S_ + s) * SH_ + hh) * SW_ + ww];
        if (v != 0.f) { lab = s; sv = v; break; }
    }
    size_t pix = (size_t)p * 128;   // NHWC, 128 contiguous halfs per pixel
    float bgseg = sv * bv;
    for (int c = 0; c < SD_; ++c)
        omap[pix + c] = (_Float16)(codes[((size_t)b * S_ + lab) * SD_ + c] * bgseg);
    for (int s = 0; s < S_; ++s)
        omap[pix + SD_ + s] = (_Float16)((s == lab) ? bgseg : 0.f);
    for (int c = SD_ + S_; c < 128; ++c)
        omap[pix + c] = (_Float16)0.f;
}

// ---------------------------------------------------------------------------
// Instance norm (biased var), in place; one block per (b, channel).
// ---------------------------------------------------------------------------
__global__ void __launch_bounds__(256)
inorm_kernel(float* __restrict__ dx) {
    __shared__ float rs[256], rs2[256];
    size_t base = (size_t)blockIdx.x * (H_ * W_);
    int tid = threadIdx.x;
    float s = 0.f, ss = 0.f;
    for (int p = tid; p < H_ * W_; p += 256) {
        float v = dx[base + p];
        s += v; ss += v * v;
    }
    rs[tid] = s; rs2[tid] = ss;
    __syncthreads();
    for (int off = 128; off > 0; off >>= 1) {
        if (tid < off) { rs[tid] += rs[tid + off]; rs2[tid] += rs2[tid + off]; }
        __syncthreads();
    }
    float mu  = rs[0] / (float)(H_ * W_);
    float var = rs2[0] / (float)(H_ * W_) - mu * mu;
    float r = rsqrtf(var + 1e-5f);
    for (int p = tid; p < H_ * W_; p += 256)
        dx[base + p] = (dx[base + p] - mu) * r;
}

// ---------------------------------------------------------------------------
extern "C" void kernel_launch(void* const* d_in, const int* in_sizes, int n_in,
                              void* d_out, int out_size, void* d_ws, size_t ws_size,
                              hipStream_t stream) {
    const float* featmap_in = (const float*)d_in[0];
    const float* segmap     = (const float*)d_in[1];
    const float* bg         = (const float*)d_in[2];
    const float* mask       = (const float*)d_in[3];
    const float* conv_w     = (const float*)d_in[4];
    const float* conv_b     = (const float*)d_in[5];
    const float* conv0_w    = (const float*)d_in[6];
    const float* conv0_b    = (const float*)d_in[7];
    const float* mlp_w      = (const float*)d_in[8];
    const float* mlp_b      = (const float*)d_in[9];
    const float* gamma_w    = (const float*)d_in[10];
    const float* gamma_b    = (const float*)d_in[11];
    const float* beta_w     = (const float*)d_in[12];
    const float* beta_b     = (const float*)d_in[13];

    char* ws = (char*)d_ws;
    size_t off = 0;
    auto alloc = [&](size_t bytes) {
        char* p = ws + off;
        off = (off + bytes + 255) & ~(size_t)255;
        return p;
    };
    _Float16* w16_conv  = (_Float16*)alloc((size_t)64 * 9 * 256 * 2);
    _Float16* w16_conv0 = (_Float16*)alloc((size_t)256 * 9 * 256 * 2);
    _Float16* w16_mlp   = (_Float16*)alloc((size_t)128 * 9 * 128 * 2);
    _Float16* w16_g     = (_Float16*)alloc((size_t)256 * 9 * 128 * 2);
    _Float16* w16_bt    = (_Float16*)alloc((size_t)256 * 9 * 128 * 2);
    float*    featmap = (float*)alloc((size_t)B_ * SD_ * H_ * W_ * 4);          // f32 NCHW
    float*    codes   = (float*)alloc((size_t)B_ * S_ * SD_ * 4);
    _Float16* omap16  = (_Float16*)alloc((size_t)B_ * H_ * W_ * 128 * 2);       // f16 NHWC
    float*    dx      = (float*)alloc((size_t)B_ * CIN_ * H_ * W_ * 4);         // f32 NCHW
    _Float16* actv16  = (_Float16*)alloc((size_t)B_ * H_ * W_ * NH_ * 2);       // f16 NHWC

    auto launch_sw = [&](const float* w, _Float16* d, int Cout, int CinS, int Cinp) {
        int total = Cout * 9 * Cinp;
        swizzle_w_kernel<<<(total + 255) / 256, 256, 0, stream>>>(w, d, Cout, CinS, Cinp);
    };
    launch_sw(conv_w,  w16_conv,  64,  256, 256);
    launch_sw(conv0_w, w16_conv0, 256, 256, 256);
    launch_sw(mlp_w,   w16_mlp,   128, 99,  128);
    launch_sw(gamma_w, w16_g,     256, 128, 128);
    launch_sw(beta_w,  w16_bt,    256, 128, 128);

    const int nblk = B_ * H_ * (W_ / 16);           // 4096 workgroups
    const size_t sh256 = (size_t)3 * 18 * (256 + 8) * 2;
    const size_t sh128 = (size_t)3 * 18 * (128 + 8) * 2;

    // featmap = lrelu(conv3x3_256->64(featmap_in))        [f32 NCHW in/out]
    conv3x3_wmma<256, 64, 1, false, false, false><<<nblk, 256, sh256, stream>>>(
        featmap_in, w16_conv, nullptr, conv_b, nullptr, nullptr, featmap);
    // codes[b,s,64] masked means
    stats_kernel<<<B_ * S_, 256, 0, stream>>>(featmap, segmap, bg, mask, codes);
    // out_map (style broadcast + bg_segmap + zero pad) -> f16 NHWC
    stylemap_kernel<<<(B_ * H_ * W_ + 255) / 256, 256, 0, stream>>>(
        segmap, bg, codes, omap16);
    // dx = conv3x3_256->256(featmap_in); nx = instance_norm(dx) in place
    conv3x3_wmma<256, 256, 0, false, false, false><<<nblk, 256, sh256, stream>>>(
        featmap_in, w16_conv0, nullptr, conv0_b, nullptr, nullptr, dx);
    inorm_kernel<<<B_ * CIN_, 256, 0, stream>>>(dx);
    // actv = relu(conv3x3_128->128(out_map))  [f16 NHWC in (async) / f16 NHWC out]
    conv3x3_wmma<128, 128, 2, false, true, true><<<nblk, 256, sh128, stream>>>(
        omap16, w16_mlp, nullptr, mlp_b, nullptr, nullptr, actv16);
    // out = lrelu(nx*(1+gamma)+beta), dual-WMMA conv [f16 NHWC in (async), f32 out]
    conv3x3_wmma<128, 256, 0, true, true, false><<<nblk, 256, sh128, stream>>>(
        actv16, w16_g, w16_bt, gamma_b, beta_b, dx, (void*)d_out);
}
